// HunYuanMoEMoE_56650618635037
// MI455X (gfx1250) — compile-verified, matrix operand order checked by reading, not run on hardware
//
#include <hip/hip_runtime.h>
#include <hip/hip_bf16.h>
#include <math.h>

#define T_TOKENS 4096
#define DMODEL   2048
#define NEXP     16
#define ISH      4096
#define IMOE     2048

typedef __attribute__((ext_vector_type(16))) __bf16 v16bf;
typedef __attribute__((ext_vector_type(8)))  __bf16 v8bf;
typedef __attribute__((ext_vector_type(4)))  __bf16 v4bf;
typedef __attribute__((ext_vector_type(8)))  float  v8f;
typedef __attribute__((ext_vector_type(16))) unsigned short v16u;
typedef __attribute__((ext_vector_type(8)))  unsigned short v8u;
typedef __attribute__((ext_vector_type(4)))  float  v4f;

// 16-bit A/B fragment, 16x32 (row r = lane%16 along own major dim, K split:
// lanes 0-15 hold K = 0..7 & 16..23, lanes 16-31 hold K = 8..15 & 24..31).
// p8 already includes the per-lane khalf offset.
static __device__ __forceinline__ v16bf load_afrag(const unsigned short* p8) {
  const v8u* p = (const v8u*)p8;
  v8u lo = p[0];          // K khalf .. khalf+7
  v8u hi = p[2];          // K khalf+16 .. khalf+23
  v16u u = __builtin_shufflevector(lo, hi, 0, 1, 2, 3, 4, 5, 6, 7,
                                           8, 9, 10, 11, 12, 13, 14, 15);
  return __builtin_bit_cast(v16bf, u);
}

// Same fragment sourced from fp32 weights; native bf16 converts (RNE).
static __device__ __forceinline__ v16bf load_bfrag(const float* pf) {
  const v4f* p = (const v4f*)pf;
  v4bf b0 = __builtin_convertvector(p[0], v4bf);
  v4bf b1 = __builtin_convertvector(p[1], v4bf);
  v4bf b2 = __builtin_convertvector(p[4], v4bf);
  v4bf b3 = __builtin_convertvector(p[5], v4bf);
  v8bf lo = __builtin_shufflevector(b0, b1, 0, 1, 2, 3, 4, 5, 6, 7);
  v8bf hi = __builtin_shufflevector(b2, b3, 0, 1, 2, 3, 4, 5, 6, 7);
  return __builtin_shufflevector(lo, hi, 0, 1, 2, 3, 4, 5, 6, 7,
                                         8, 9, 10, 11, 12, 13, 14, 15);
}

// Generic C[m,n] = sum_k A[m,k]*W[n,k].  A bf16 row-major [*, lda], W fp32
// row-major [N, ldw] (per-expert stride wstride when offsets != null).
// Routed mode (offsets != null): e = blockIdx.z, rows = offsets[e+1]-offsets[e],
// A row = elist[rbase+r] (gatherA) or rbase+r (compact), C row = rbase+r.
// Block = 256 threads = 8 waves (2 M x 4 N); wave tile 128x64 -> block 256x256.
__global__ __launch_bounds__(256)
void moe_gemm_bf16_kernel(const unsigned short* __restrict__ A, int lda,
                          const float* __restrict__ W, unsigned long long wstride, int ldw,
                          float* __restrict__ C, int ldc,
                          int Mfixed, int K,
                          const int* __restrict__ offsets,
                          const int* __restrict__ elist,
                          int gatherA) {
  const int lane = threadIdx.x & 31;
  const int wid  = threadIdx.x >> 5;
  const int wm   = wid & 1;
  const int wn   = wid >> 1;

  int rbase = 0, M = Mfixed;
  const float* Wp = W;
  if (offsets) {
    const int z = blockIdx.z;
    rbase = offsets[z];
    M = offsets[z + 1] - rbase;
    Wp = W + (size_t)z * wstride;
  }
  if ((int)blockIdx.x * 256 >= M) return;   // uniform early-exit for empty tiles

  const int Mbase = blockIdx.x * 256 + wm * 128;
  const int Nbase = blockIdx.y * 256 + wn * 64;
  const int lrow  = lane & 15;
  const int khalf = (lane >= 16) ? 8 : 0;

  const unsigned short* aptr[8];
#pragma unroll
  for (int i = 0; i < 8; ++i) {
    int r  = Mbase + i * 16 + lrow;
    int rr = (r < M) ? r : (M - 1);      // clamp ragged rows; masked at store
    int arow = gatherA ? elist[rbase + rr] : (rbase + rr);
    aptr[i] = A + (size_t)arow * lda + khalf;
  }
  const float* bptr[4];
#pragma unroll
  for (int j = 0; j < 4; ++j) {
    int n = Nbase + j * 16 + lrow;       // N is always a multiple of 256 here
    bptr[j] = Wp + (size_t)n * ldw + khalf;
  }

  v8f acc[8][4] = {};

  for (int kt = 0; kt < K; kt += 32) {
    v16bf a[8];
#pragma unroll
    for (int i = 0; i < 8; ++i) a[i] = load_afrag(aptr[i] + kt);
#pragma unroll
    for (int j = 0; j < 4; ++j) {
      v16bf b = load_bfrag(bptr[j] + kt);
      __builtin_prefetch(bptr[j] + kt + 32, 0, 1);   // next K-step of weights
#pragma unroll
      for (int i = 0; i < 8; ++i) {
        acc[i][j] = __builtin_amdgcn_wmma_f32_16x16x32_bf16(
            false, a[i], false, b, (short)0, acc[i][j], false, false);
      }
    }
  }

  // C/D layout: elem v, lanes 0-15 -> M=v, lanes 16-31 -> M=v+8; N = lane%16
  const int mhalf = (lane >= 16) ? 8 : 0;
#pragma unroll
  for (int i = 0; i < 8; ++i)
#pragma unroll
    for (int j = 0; j < 4; ++j)
#pragma unroll
      for (int v = 0; v < 8; ++v) {
        int m = Mbase + i * 16 + v + mhalf;
        if (m < M)
          C[(size_t)(rbase + m) * ldc + (Nbase + j * 16 + lrow)] = acc[i][j][v];
      }
}

// fp32 -> bf16 activation conversion
__global__ void moe_cvt_kernel(const float* __restrict__ x,
                               __bf16* __restrict__ x16, long n) {
  long i = (long)blockIdx.x * blockDim.x + threadIdx.x;
  if (i < n) x16[i] = (__bf16)x[i];
}

// Gating: 16 fp32 logits per token, top-2, renormalized softmax weights.
__global__ __launch_bounds__(128)
void moe_gate_kernel(const float* __restrict__ x, const float* __restrict__ wg,
                     int* __restrict__ topk_idx, float* __restrict__ topk_w) {
  const int t = blockIdx.x;
  const int tid = threadIdx.x;
  const int lane = tid & 31;
  const int wid = tid >> 5;
  float acc[NEXP];
#pragma unroll
  for (int e = 0; e < NEXP; ++e) acc[e] = 0.f;
  const float* xr = x + (size_t)t * DMODEL;
  for (int k = tid; k < DMODEL; k += 128) {
    float xv = xr[k];
#pragma unroll
    for (int e = 0; e < NEXP; ++e) acc[e] += xv * wg[e * DMODEL + k];
  }
  __shared__ float sred[4][NEXP];
#pragma unroll
  for (int e = 0; e < NEXP; ++e) {
    float v = acc[e];
#pragma unroll
    for (int off = 16; off >= 1; off >>= 1) v += __shfl_xor(v, off, 32);
    if (lane == 0) sred[wid][e] = v;
  }
  __syncthreads();
  if (tid == 0) {
    float lg[NEXP];
#pragma unroll
    for (int e = 0; e < NEXP; ++e)
      lg[e] = sred[0][e] + sred[1][e] + sred[2][e] + sred[3][e];
    int i0 = 0; float l0 = lg[0];
    for (int e = 1; e < NEXP; ++e) if (lg[e] > l0) { l0 = lg[e]; i0 = e; }
    int i1 = (i0 == 0) ? 1 : 0; float l1 = lg[i1];
    for (int e = 0; e < NEXP; ++e)
      if (e != i0 && lg[e] > l1) { l1 = lg[e]; i1 = e; }
    // full-softmax denominator cancels under top-k renormalization
    float w0 = 1.f / (1.f + expf(l1 - l0));
    topk_idx[t * 2]     = i0;
    topk_idx[t * 2 + 1] = i1;
    topk_w[t * 2]       = w0;
    topk_w[t * 2 + 1]   = 1.f - w0;
  }
}

// Per-expert counts + exclusive prefix (single block; also resets cursors).
__global__ void moe_count_kernel(const int* __restrict__ topk_idx,
                                 int* __restrict__ offsets,
                                 int* __restrict__ cursor) {
  __shared__ int sc[NEXP];
  int tid = threadIdx.x;
  if (tid < NEXP) sc[tid] = 0;
  __syncthreads();
  for (int i = tid; i < T_TOKENS * 2; i += blockDim.x)
    atomicAdd(&sc[topk_idx[i]], 1);
  __syncthreads();
  if (tid == 0) {
    int run = 0;
    for (int e = 0; e < NEXP; ++e) { offsets[e] = run; cursor[e] = run; run += sc[e]; }
    offsets[NEXP] = run;   // == 2*T
  }
}

// Scatter token ids into per-expert compact lists; record each slot's row.
// (List order is atomic-nondeterministic, but every GEMM row depends only on
//  its own token id and the combine sums slots in fixed order -> deterministic.)
__global__ void moe_fill_kernel(const int* __restrict__ topk_idx,
                                int* __restrict__ cursor,
                                int* __restrict__ elist,
                                int* __restrict__ slot_row) {
  int i = blockIdx.x * blockDim.x + threadIdx.x;
  if (i < T_TOKENS * 2) {
    int e = topk_idx[i];
    int p = atomicAdd(&cursor[e], 1);
    elist[p] = i >> 1;
    slot_row[i] = p;
  }
}

// h = bf16( silu(a) * b ); routed variant bounds by offsets[16]*cols.
__global__ void moe_silu_mul_kernel(const float* __restrict__ a,
                                    const float* __restrict__ b,
                                    __bf16* __restrict__ h,
                                    long total_fixed,
                                    const int* __restrict__ offsets, int cols) {
  long i = (long)blockIdx.x * blockDim.x + threadIdx.x;
  long total = offsets ? (long)offsets[NEXP] * cols : total_fixed;
  if (i < total) {
    float av = a[i];
    float s = av / (1.f + expf(-av));
    h[i] = (__bf16)(s * b[i]);
  }
}

// out[t,:] += w0*y[slot0,:] + w1*y[slot1,:]   (fixed summation order)
__global__ void moe_combine_kernel(float* __restrict__ out,
                                   const float* __restrict__ y,
                                   const int* __restrict__ slot_row,
                                   const float* __restrict__ topk_w) {
  long i = (long)blockIdx.x * blockDim.x + threadIdx.x;
  if (i < (long)T_TOKENS * DMODEL) {
    int t = (int)(i >> 11);               // D = 2048
    int d = (int)(i & (DMODEL - 1));
    int r0 = slot_row[t * 2], r1 = slot_row[t * 2 + 1];
    out[i] += topk_w[t * 2]     * y[(size_t)r0 * DMODEL + d]
            + topk_w[t * 2 + 1] * y[(size_t)r1 * DMODEL + d];
  }
}

extern "C" void kernel_launch(void* const* d_in, const int* in_sizes, int n_in,
                              void* d_out, int out_size, void* d_ws, size_t ws_size,
                              hipStream_t stream) {
  (void)in_sizes; (void)n_in; (void)out_size; (void)ws_size;
  const float* x   = (const float*)d_in[0];
  const float* wg  = (const float*)d_in[1];
  const float* sw1 = (const float*)d_in[2];
  const float* sw3 = (const float*)d_in[3];
  const float* sw2 = (const float*)d_in[4];
  const float* ew1 = (const float*)d_in[5];
  const float* ew3 = (const float*)d_in[6];
  const float* ew2 = (const float*)d_in[7];
  float* out = (float*)d_out;

  const size_t MB = 1024ull * 1024ull;
  char* ws = (char*)d_ws;
  __bf16* x16 = (__bf16*)(ws);                           // 16 MB  [4096 x 2048] bf16
  float* t1 = (float*)(ws + 16 * MB);                    // 64 MB  16M fp32
  float* t3 = (float*)(ws + 80 * MB);                    // 64 MB  16M fp32
  __bf16* h = (__bf16*)(ws + 144 * MB);                  // 32 MB  16M bf16
  float* y = t1;                                         // reuse t1 for routed y
  char* meta = ws + 176 * MB;
  int*   topk_idx = (int*)(meta);                        // 8192 ints
  float* topk_w   = (float*)(meta + 64 * 1024);          // 8192 floats
  int*   offsets  = (int*)(meta + 128 * 1024);           // 17 ints
  int*   cursor   = (int*)(meta + 128 * 1024 + 256);     // 16 ints
  int*   elist    = (int*)(meta + 192 * 1024);           // 8192 ints
  int*   slot_row = (int*)(meta + 256 * 1024);           // 8192 ints

  // 1. activations -> bf16 (weights stay fp32 in HBM; converted in-register)
  {
    long n = (long)T_TOKENS * DMODEL;
    moe_cvt_kernel<<<dim3((unsigned)((n + 255) / 256)), dim3(256), 0, stream>>>(x, x16, n);
  }
  // 2. gate + top-2
  moe_gate_kernel<<<dim3(T_TOKENS), dim3(128), 0, stream>>>(x, wg, topk_idx, topk_w);
  // 3. routing tables
  moe_count_kernel<<<dim3(1), dim3(256), 0, stream>>>(topk_idx, offsets, cursor);
  moe_fill_kernel<<<dim3((T_TOKENS * 2 + 255) / 256), dim3(256), 0, stream>>>(
      topk_idx, cursor, elist, slot_row);

  // 4. shared expert: t1 = x @ sw1^T, t3 = x @ sw3^T   [4096 x 4096]
  {
    dim3 grid(T_TOKENS / 256, ISH / 256, 1);
    moe_gemm_bf16_kernel<<<grid, dim3(256), 0, stream>>>(
        (const unsigned short*)x16, DMODEL, sw1, 0ull, DMODEL,
        t1, ISH, T_TOKENS, DMODEL, nullptr, nullptr, 0);
    moe_gemm_bf16_kernel<<<grid, dim3(256), 0, stream>>>(
        (const unsigned short*)x16, DMODEL, sw3, 0ull, DMODEL,
        t3, ISH, T_TOKENS, DMODEL, nullptr, nullptr, 0);
  }
  // 5. h = silu(t1) * t3   [4096 x 4096] bf16
  {
    long n = (long)T_TOKENS * ISH;
    moe_silu_mul_kernel<<<dim3((unsigned)((n + 255) / 256)), dim3(256), 0, stream>>>(
        t1, t3, h, n, nullptr, ISH);
  }
  // 6. out = h @ sw2^T   [4096 x 2048]  (writes every element of d_out)
  {
    dim3 grid(T_TOKENS / 256, DMODEL / 256, 1);
    moe_gemm_bf16_kernel<<<grid, dim3(256), 0, stream>>>(
        (const unsigned short*)h, ISH, sw2, 0ull, ISH,
        out, DMODEL, T_TOKENS, ISH, nullptr, nullptr, 0);
  }
  // 7. routed w1/w3 with row gather (compact per-expert outputs in t1/t3)
  {
    dim3 grid(T_TOKENS / 256, IMOE / 256, NEXP);
    unsigned long long wstr = (unsigned long long)IMOE * DMODEL;
    moe_gemm_bf16_kernel<<<grid, dim3(256), 0, stream>>>(
        (const unsigned short*)x16, DMODEL, ew1, wstr, DMODEL,
        t1, IMOE, 0, DMODEL, offsets, elist, 1);
    moe_gemm_bf16_kernel<<<grid, dim3(256), 0, stream>>>(
        (const unsigned short*)x16, DMODEL, ew3, wstr, DMODEL,
        t3, IMOE, 0, DMODEL, offsets, elist, 1);
  }
  // 8. routed h = silu(t1)*t3   [<=8192 x 2048] bf16
  {
    long n = (long)T_TOKENS * 2 * IMOE;   // worst-case grid; device clamps
    moe_silu_mul_kernel<<<dim3((unsigned)((n + 255) / 256)), dim3(256), 0, stream>>>(
        t1, t3, h, n, offsets, IMOE);
  }
  // 9. y = h @ ew2^T per expert (compact rows; y aliases t1, which is dead)
  {
    dim3 grid(T_TOKENS / 256, DMODEL / 256, NEXP);
    unsigned long long wstr = (unsigned long long)DMODEL * IMOE;
    moe_gemm_bf16_kernel<<<grid, dim3(256), 0, stream>>>(
        (const unsigned short*)h, IMOE, ew2, wstr, IMOE,
        y, DMODEL, 0, IMOE, offsets, elist, 0);
  }
  // 10. out += w0*y[slot0] + w1*y[slot1]
  {
    long n = (long)T_TOKENS * DMODEL;
    moe_combine_kernel<<<dim3((unsigned)((n + 255) / 256)), dim3(256), 0, stream>>>(
        out, y, slot_row, topk_w);
  }
}
